// top_k_quantization_23304492548628
// MI455X (gfx1250) — matrix operand order repaired and checked by reading, not compiled, not run on hardware
//
#include <hip/hip_runtime.h>
#include <hip/hip_bf16.h>

typedef __attribute__((ext_vector_type(16))) __bf16 v16bf;
typedef __attribute__((ext_vector_type(8)))  __bf16 v8bf;
typedef __attribute__((ext_vector_type(8)))  float  v8f;

#define LRELU_SLOPE 0.2f

// ---------------------------------------------------------------------------
// Implicit-GEMM conv via WMMA bf16 (f32 accumulate), 64M x 128N block tile,
// double-buffered LDS pipeline (1 barrier / K-step, staging overlaps WMMA).
//   K ordering: k = khw*Cin + ci  (khw = kh*ks+kw), Cin a power of two.
//   A: staged 64M x 32K row-major bf16 (fragment K contiguous).
//   B: staged transposed 128N x 32K bf16 (fragment K contiguous).
//   4 waves; wave wv computes M rows [wv*16,wv*16+16) x 128 N cols = 8 WMMAs/step.
// ---------------------------------------------------------------------------
__global__ __launch_bounds__(128)
void vq_conv_wmma_kernel(const float* __restrict__ in0,
                         const float* __restrict__ in1,   // concat second half (may be null)
                         const float* __restrict__ wgt,
                         const float* __restrict__ bias,
                         float* __restrict__ out,
                         int Nb, int Cin, int lgCin, int Csplit, int Hin, int Win,
                         int Cout, int Hout, int Wout,
                         int ks, int stride, int pad, int tmode, float alpha)
{
    __shared__ __align__(32) __bf16 As[2][64 * 32];
    __shared__ __align__(32) __bf16 Bs[2][128 * 32];

    const int tid  = threadIdx.x;
    const int lane = tid & 31;
    const int wv   = tid >> 5;
    const int M    = Nb * Hout * Wout;
    const int ks2  = ks * ks;
    const int Ktot = Cin * ks2;
    const int ciMask = Cin - 1;
    const int m0   = blockIdx.x * 64;
    const int n0   = blockIdx.y * 128;
    const int HWo  = Hout * Wout;
    const int HWin = Hin * Win;
    const int Cin1 = Cin - Csplit;

    // ---- per-thread staging coordinates (fixed across K loop) ----
    const int aml  = tid & 63;            // A: row in tile
    const int aklb = (tid >> 6) * 16;     // A: K half
    int am = m0 + aml; if (am >= M) am = M - 1;
    const int anb = am / HWo;
    const int arp = am - anb * HWo;
    const int aoh = arp / Wout;
    const int aow = arp - aoh * Wout;
    const float* aBase0 = in0 + (anb * Csplit) * HWin;
    const float* aBase1 = (in1 != nullptr) ? (in1 + (anb * Cin1) * HWin) : nullptr;

    const int bnl  = tid;                 // B: column, all 32 K slots
    const int bco  = n0 + bnl;
    const bool bvalid = (bco < Cout);

    v8f acc[8];
#pragma unroll
    for (int j = 0; j < 8; ++j)
#pragma unroll
        for (int r = 0; r < 8; ++r) acc[j][r] = 0.0f;

    auto mma_step = [&](int buf) {
        const int mrow = lane & 15;
        const int kbA  = (lane < 16) ? 0 : 16;
        union { v16bf v; v8bf h[2]; } ua;
        const __bf16* arow = &As[buf][(wv * 16 + mrow) * 32];
        ua.h[0] = *(const v8bf*)(arow + kbA);
        ua.h[1] = *(const v8bf*)(arow + kbA + 8);
        const int ncol = lane & 15;
        const int kbB  = (lane < 16) ? 0 : 16;
#pragma unroll
        for (int j = 0; j < 8; ++j) {
            v16bf b = *(const v16bf*)(&Bs[buf][(j * 16 + ncol) * 32 + kbB]);
            acc[j] = __builtin_amdgcn_wmma_f32_16x16x32_bf16(
                false, ua.v, false, b, (short)0, acc[j], false, false);
        }
    };

    if (Cin >= 32) {
        // =============== fast path: uniform (kh,kw) per 32-chunk ===============
        const int lgCh   = lgCin - 5;            // chunks per khw = Cin/32
        const int chMask = (1 << lgCh) - 1;
        const int nsteps = ks2 << lgCh;

        auto stage_fast = [&](int step, int buf) {
            unsigned int* As32 = (unsigned int*)&As[buf][0];
            unsigned int* Bs32 = (unsigned int*)&Bs[buf][0];
            const int khw = step >> lgCh;
            const int cb  = (step & chMask) << 5;
            const int kh  = (khw * 11) >> 5;     // exact /3 for khw in [0,8]
            const int kw  = khw - kh * 3;
            int ih, iw; bool aok;
            if (tmode == 0) {
                ih = aoh * stride - pad + kh;
                iw = aow * stride - pad + kw;
                aok = (ih >= 0) & (ih < Hin) & (iw >= 0) & (iw < Win);
            } else {                              // conv-transpose s2/p1/op1
                int dh = aoh + kh - 1, dw = aow + kw - 1;
                ih = dh >> 1; iw = dw >> 1;
                aok = (dh >= 0) & (dw >= 0) & ((dh & 1) == 0) & ((dw & 1) == 0) &
                      (ih < Hin) & (iw < Win);
            }
            const int aoff = ih * Win + iw;
            // ---- stage A: strided channel loads ----
            {
                const float* src = (cb < Csplit)
                    ? (aBase0 + (cb + aklb) * HWin)
                    : (aBase1 + (cb - Csplit + aklb) * HWin);
                if (aok) {
                    const float* sp = src + aoff;
                    __builtin_prefetch(sp + 32 * HWin, 0, 1);
#pragma unroll
                    for (int u = 0; u < 8; ++u) {
                        union { __bf16 h[2]; unsigned int u32; } p;
                        p.h[0] = (__bf16)sp[(2 * u) * HWin];
                        p.h[1] = (__bf16)sp[(2 * u + 1) * HWin];
                        As32[(aml * 32 + aklb + 2 * u) >> 1] = p.u32;
                    }
                } else {
#pragma unroll
                    for (int u = 0; u < 8; ++u)
                        As32[(aml * 32 + aklb + 2 * u) >> 1] = 0u;
                }
            }
            // ---- stage B: strided weight loads, Bs[n][k] ----
            if (bvalid) {
                if (tmode == 0) {
                    const float* wb = wgt + (bco * Cin + cb) * ks2 + khw;
#pragma unroll
                    for (int u = 0; u < 16; ++u) {
                        union { __bf16 h[2]; unsigned int u32; } p;
                        p.h[0] = (__bf16)wb[(2 * u) * ks2];
                        p.h[1] = (__bf16)wb[(2 * u + 1) * ks2];
                        Bs32[bnl * 16 + u] = p.u32;
                    }
                } else {
                    const float* wb = wgt + (cb * Cout + bco) * ks2 + (ks2 - 1 - khw);
                    const int wstr = Cout * ks2;
#pragma unroll
                    for (int u = 0; u < 16; ++u) {
                        union { __bf16 h[2]; unsigned int u32; } p;
                        p.h[0] = (__bf16)wb[(2 * u) * wstr];
                        p.h[1] = (__bf16)wb[(2 * u + 1) * wstr];
                        Bs32[bnl * 16 + u] = p.u32;
                    }
                }
            } else {
#pragma unroll
                for (int u = 0; u < 16; ++u) Bs32[bnl * 16 + u] = 0u;
            }
        };

        stage_fast(0, 0);
        for (int s = 0; s < nsteps; ++s) {
            __syncthreads();                      // buf s&1 ready; prev reads done
            if (s + 1 < nsteps) stage_fast(s + 1, (s + 1) & 1);
            mma_step(s & 1);
        }
    } else {
        // =============== generic path (tiny Cin, e0 only) ===============
        unsigned int* As32 = (unsigned int*)&As[0][0];
        unsigned int* Bs32 = (unsigned int*)&Bs[0][0];
        auto gatherA = [&](int k) -> float {
            if (k >= Ktot) return 0.0f;
            int ci  = k & ciMask;
            int khw = k >> lgCin;
            int kh  = (khw * 11) >> 5;
            int kw  = khw - kh * 3;
            int ih = aoh * stride - pad + kh;
            int iw = aow * stride - pad + kw;
            if (ih < 0 || ih >= Hin || iw < 0 || iw >= Win) return 0.0f;
            return aBase0[ci * HWin + ih * Win + iw];
        };
        auto gatherB = [&](int k) -> float {
            if (k >= Ktot || !bvalid) return 0.0f;
            int ci  = k & ciMask;
            int khw = k >> lgCin;
            return wgt[(bco * Cin + ci) * ks2 + khw];
        };
        for (int kk = 0; kk < Ktot; kk += 32) {
            __syncthreads();
#pragma unroll
            for (int u = 0; u < 8; ++u) {
                union { __bf16 h[2]; unsigned int u32; } p;
                p.h[0] = (__bf16)gatherA(kk + aklb + 2 * u);
                p.h[1] = (__bf16)gatherA(kk + aklb + 2 * u + 1);
                As32[(aml * 32 + aklb + 2 * u) >> 1] = p.u32;
            }
#pragma unroll
            for (int u = 0; u < 16; ++u) {
                union { __bf16 h[2]; unsigned int u32; } p;
                p.h[0] = (__bf16)gatherB(kk + 2 * u);
                p.h[1] = (__bf16)gatherB(kk + 2 * u + 1);
                Bs32[bnl * 16 + u] = p.u32;
            }
            __syncthreads();
            mma_step(0);
        }
    }

    // ---- epilogue: alpha*acc + bias, scatter to NCHW ----
    const int ncol = lane & 15;
    float bb[8];
#pragma unroll
    for (int j = 0; j < 8; ++j) {
        int co = n0 + j * 16 + ncol;
        bb[j] = (bias != nullptr && co < Cout) ? bias[co] : 0.0f;
    }
#pragma unroll
    for (int r = 0; r < 8; ++r) {
        int mr = m0 + wv * 16 + r + ((lane < 16) ? 0 : 8);
        if (mr < M) {
            int nb = mr / HWo;
            int rp = mr - nb * HWo;
            int oh = rp / Wout;
            int ow = rp - oh * Wout;
            float* orow = out + (nb * Cout) * HWo + oh * Wout + ow;
#pragma unroll
            for (int j = 0; j < 8; ++j) {
                int co = n0 + j * 16 + ncol;
                if (co < Cout)
                    orow[co * HWo] = alpha * acc[j][r] + bb[j];
            }
        }
    }
}

// ---------------------------------------------------------------------------
// GroupNorm stats: slab partial sums -> atomicAdd into sums[2*(n*groups+gr)].
// Grid: (N*groups*slabs) blocks. count must divide evenly by slabs.
// ---------------------------------------------------------------------------
__global__ __launch_bounds__(256)
void vq_gn_stats_kernel(const float* __restrict__ x, float* __restrict__ sums,
                        int Cg, int HW, int slabs)
{
    __shared__ float r1[256];
    __shared__ float r2[256];
    const int bg = blockIdx.x / slabs;        // n*groups + gr
    const int sl = blockIdx.x % slabs;
    const int count = Cg * HW;
    const int per   = count / slabs;
    const int base  = bg * count + sl * per;

    float s = 0.0f, s2 = 0.0f;
    for (int i = threadIdx.x; i < per; i += 256) {
        float v = x[base + i];
        s += v; s2 += v * v;
    }
    r1[threadIdx.x] = s; r2[threadIdx.x] = s2;
    __syncthreads();
    for (int off = 128; off > 0; off >>= 1) {
        if (threadIdx.x < off) {
            r1[threadIdx.x] += r1[threadIdx.x + off];
            r2[threadIdx.x] += r2[threadIdx.x + off];
        }
        __syncthreads();
    }
    if (threadIdx.x == 0) {
        atomicAdd(&sums[2 * bg + 0], r1[0]);
        atomicAdd(&sums[2 * bg + 1], r2[0]);
    }
}

// ---------------------------------------------------------------------------
// GroupNorm apply + LeakyReLU, element-wise over the whole tensor.
// ---------------------------------------------------------------------------
__global__ __launch_bounds__(256)
void vq_gn_apply_kernel(const float* __restrict__ x, const float* __restrict__ sums,
                        const float* __restrict__ g, const float* __restrict__ beta,
                        float* __restrict__ y, int C, int HW, int groups, int total)
{
    int t = blockIdx.x * 256 + threadIdx.x;
    if (t >= total) return;
    const int chw = C * HW;
    const int n   = t / chw;
    const int rem = t - n * chw;
    const int c   = rem / HW;
    const int Cg  = C / groups;
    const int gr  = c / Cg;
    const int bg  = n * groups + gr;
    const float cnt  = (float)(Cg * HW);
    const float mean = sums[2 * bg + 0] / cnt;
    const float var  = sums[2 * bg + 1] / cnt - mean * mean;
    const float rs   = rsqrtf(var + 1e-5f);
    float o = (x[t] - mean) * rs * g[c] + beta[c];
    y[t] = (o >= 0.0f) ? o : LRELU_SLOPE * o;
}

__global__ __launch_bounds__(256)
void vq_code_norm_kernel(const float* __restrict__ emb, float* __restrict__ norms,
                         int ncodes, int dim)
{
    int c = blockIdx.x * 256 + threadIdx.x;
    if (c < ncodes) {
        float s = 0.0f;
        for (int k = 0; k < dim; ++k) { float e = emb[c * dim + k]; s += e * e; }
        norms[c] = s;
    }
}

__global__ __launch_bounds__(256)
void vq_zero_kernel(float* __restrict__ p, int n)
{
    int i = blockIdx.x * 256 + threadIdx.x;
    if (i < n) p[i] = 0.0f;
}

// ---------------------------------------------------------------------------
// Per-row top-3 smallest over ncodes scores. Thread per row: lanes read
// consecutive pixels of one code plane -> fully coalesced.
// ---------------------------------------------------------------------------
__global__ __launch_bounds__(256)
void vq_top3_kernel(const float* __restrict__ scores, int* __restrict__ idx0,
                    int* __restrict__ idx2, int rows, int ncodes, int HW)
{
    int t = blockIdx.x * 256 + threadIdx.x;
    if (t >= rows) return;
    const int nb = t / HW;
    const int p  = t - nb * HW;
    const float* base = scores + (nb * ncodes) * HW + p;

    float d0 = 3.0e38f, d1 = 3.0e38f, d2 = 3.0e38f;
    int   i0 = 0, i1 = 0, i2 = 0;
    for (int c = 0; c < ncodes; ++c) {
        float d = base[c * HW];
        if (d < d0)      { d2 = d1; i2 = i1; d1 = d0; i1 = i0; d0 = d; i0 = c; }
        else if (d < d1) { d2 = d1; i2 = i1; d1 = d;  i1 = c; }
        else if (d < d2) { d2 = d;  i2 = c; }
    }
    idx0[t] = i0;
    idx2[t] = i2;
}

// ---------------------------------------------------------------------------
// quantized = emb[idx0[row]] in NCHW order (z/q coalesced, emb L2 gather);
// accumulate sum((q-z)^2).
// ---------------------------------------------------------------------------
__global__ __launch_bounds__(256)
void vq_quant_kernel(const float* __restrict__ z, const float* __restrict__ emb,
                     const int* __restrict__ idx0, float* __restrict__ q,
                     float* __restrict__ lossAcc, int Nb, int Cdim, int HW)
{
    __shared__ float red[256];
    const int total = Nb * Cdim * HW;
    int t = blockIdx.x * 256 + threadIdx.x;
    float d2 = 0.0f;
    if (t < total) {
        int nb  = t / (Cdim * HW);
        int rem = t - nb * (Cdim * HW);
        int c   = rem / HW;
        int p   = rem - c * HW;
        int r   = nb * HW + p;
        float zv = z[t];
        float qv = emb[idx0[r] * Cdim + c];
        q[t] = qv;
        float df = qv - zv;
        d2 = df * df;
    }
    red[threadIdx.x] = d2;
    __syncthreads();
    for (int off = 128; off > 0; off >>= 1) {
        if (threadIdx.x < off) red[threadIdx.x] += red[threadIdx.x + off];
        __syncthreads();
    }
    if (threadIdx.x == 0) atomicAdd(lossAcc, red[0]);
}

__global__ __launch_bounds__(256)
void vq_count_kernel(const int* __restrict__ idx2, float* __restrict__ counts, int rows)
{
    int t = blockIdx.x * 256 + threadIdx.x;
    if (t < rows) atomicAdd(&counts[idx2[t]], 1.0f);
}

__global__ __launch_bounds__(256)
void vq_finalize_kernel(const float* __restrict__ counts, const float* __restrict__ lossAcc,
                        float* __restrict__ outLoss, float* __restrict__ outPerp,
                        int ncodes, float invRows, float lossScale)
{
    __shared__ float red[256];
    float s = 0.0f;
    for (int c = threadIdx.x; c < ncodes; c += 256) {
        float avg = counts[c] * invRows;
        s += avg * logf(avg + 1e-10f);
    }
    red[threadIdx.x] = s;
    __syncthreads();
    for (int off = 128; off > 0; off >>= 1) {
        if (threadIdx.x < off) red[threadIdx.x] += red[threadIdx.x + off];
        __syncthreads();
    }
    if (threadIdx.x == 0) {
        *outPerp = expf(-red[0]);
        *outLoss = lossAcc[0] * lossScale;
    }
}

// ---------------------------------------------------------------------------
// Host orchestration.
// ---------------------------------------------------------------------------
static inline int ilog2i(int v) { int l = 0; while ((1 << l) < v) ++l; return l; }

static inline void launch_conv(hipStream_t st, const float* in0, const float* in1,
                               const float* w, const float* b, float* out,
                               int Nb, int Cin, int Csplit, int Hin, int Win,
                               int Cout, int Hout, int Wout,
                               int ks, int stride, int pad, int tmode, float alpha)
{
    int M = Nb * Hout * Wout;
    dim3 grid((M + 63) / 64, (Cout + 127) / 128);
    vq_conv_wmma_kernel<<<grid, 128, 0, st>>>(in0, in1, w, b, out,
        Nb, Cin, ilog2i(Cin), Csplit, Hin, Win, Cout, Hout, Wout,
        ks, stride, pad, tmode, alpha);
}

static inline void launch_gn(hipStream_t st, const float* x, float* sums,
                             const float* g, const float* be, float* y,
                             int Nb, int C, int HW)
{
    const int groups = 2, slabs = 32;
    const int total = Nb * C * HW;
    vq_gn_stats_kernel<<<Nb * groups * slabs, 256, 0, st>>>(x, sums, C / groups, HW, slabs);
    vq_gn_apply_kernel<<<(total + 255) / 256, 256, 0, st>>>(x, sums, g, be, y, C, HW, groups, total);
}

extern "C" void kernel_launch(void* const* d_in, const int* in_sizes, int n_in,
                              void* d_out, int out_size, void* d_ws, size_t ws_size,
                              hipStream_t stream)
{
    (void)in_sizes; (void)n_in; (void)out_size; (void)ws_size;

    const float* x      = (const float*)d_in[0];
    const float* e0_w   = (const float*)d_in[1];
    const float* e0_b   = (const float*)d_in[2];
    const float* e0_g   = (const float*)d_in[3];
    const float* e0_be  = (const float*)d_in[4];
    const float* e1_w   = (const float*)d_in[5];
    const float* e1_b   = (const float*)d_in[6];
    const float* e1_g   = (const float*)d_in[7];
    const float* e1_be  = (const float*)d_in[8];
    const float* pv_w   = (const float*)d_in[9];
    const float* pv_b   = (const float*)d_in[10];
    const float* emb    = (const float*)d_in[11];
    const float* d0_w   = (const float*)d_in[12];
    const float* d0_b   = (const float*)d_in[13];
    const float* d0_g   = (const float*)d_in[14];
    const float* d0_be  = (const float*)d_in[15];
    const float* d1_w   = (const float*)d_in[16];
    const float* d1_b   = (const float*)d_in[17];
    const float* d1_g   = (const float*)d_in[18];
    const float* d1_be  = (const float*)d_in[19];
    const float* proj_w = (const float*)d_in[20];
    const float* proj_b = (const float*)d_in[21];
    const float* ro_w   = (const float*)d_in[22];
    const float* ro_b   = (const float*)d_in[23];

    const int Nb = 4, C = 128, H = 96, W = 96, Hl = 48, Wl = 48;
    const int ncodes = 1024, rows = Nb * Hl * Wl;              // 9216
    const size_t big   = (size_t)Nb * C * H * W;               // 4,718,592
    const size_t small = (size_t)Nb * C * Hl * Wl;             // 1,179,648
    const size_t scsz  = (size_t)Nb * ncodes * Hl * Wl;        // 9,437,184

    float* ws = (float*)d_ws;
    size_t off = 0;
    float* enc1   = ws + off; off += big;
    float* tmp    = ws + off; off += big;
    float* latent = ws + off; off += small;
    float* z      = ws + off; off += small;
    float* scores = ws + off; off += scsz;
    float* quant  = ws + off; off += small;
    float* skip   = ws + off; off += big;
    float* hbuf   = ws + off; off += big;
    float* ybuf   = ws + off; off += big;
    float* norms  = ws + off; off += ncodes;
    // zeroed region: counts(1024) + lossA(16) + gn sums 4 layers * 16
    float* counts = ws + off; off += ncodes;
    float* lossA  = ws + off; off += 16;
    float* gnsum0 = ws + off; off += 16;
    float* gnsum1 = ws + off; off += 16;
    float* gnsum2 = ws + off; off += 16;
    float* gnsum3 = ws + off; off += 16;
    int*   idx0   = (int*)(ws + off); off += rows;
    int*   idx2   = (int*)(ws + off); off += rows;

    float* out  = (float*)d_out;               // [loss, x_recon(147456), perplexity]
    float* xrec = out + 1;

    const int nzero = ncodes + 16 + 4 * 16;
    vq_zero_kernel<<<(nzero + 255) / 256, 256, 0, stream>>>(counts, nzero);
    vq_code_norm_kernel<<<(ncodes + 255) / 256, 256, 0, stream>>>(emb, norms, ncodes, C);

    // encoder
    launch_conv(stream, x, nullptr, e0_w, e0_b, tmp, Nb, 4, 4, H, W, C, H, W, 3, 1, 1, 0, 1.0f);
    launch_gn(stream, tmp, gnsum0, e0_g, e0_be, enc1, Nb, C, H * W);

    launch_conv(stream, enc1, nullptr, e1_w, e1_b, tmp, Nb, C, C, H, W, C, Hl, Wl, 3, 2, 1, 0, 1.0f);
    launch_gn(stream, tmp, gnsum1, e1_g, e1_be, latent, Nb, C, Hl * Wl);

    launch_conv(stream, latent, nullptr, pv_w, pv_b, z, Nb, C, C, Hl, Wl, C, Hl, Wl, 1, 1, 0, 0, 1.0f);

    // VQ: scores = -2 z.e + ||e||^2  (== d - ||z||^2, same ranking)
    launch_conv(stream, z, nullptr, emb, norms, scores, Nb, C, C, Hl, Wl, ncodes, Hl, Wl, 1, 1, 0, 0, -2.0f);
    vq_top3_kernel<<<(rows + 255) / 256, 256, 0, stream>>>(scores, idx0, idx2, rows, ncodes, Hl * Wl);
    vq_quant_kernel<<<((int)small + 255) / 256, 256, 0, stream>>>(z, emb, idx0, quant, lossA, Nb, C, Hl * Wl);
    vq_count_kernel<<<(rows + 255) / 256, 256, 0, stream>>>(idx2, counts, rows);

    // skip path
    launch_conv(stream, enc1, nullptr, proj_w, proj_b, skip, Nb, C, C, H, W, C, H, W, 1, 1, 0, 0, 1.0f);

    // decoder
    launch_conv(stream, quant, nullptr, d0_w, d0_b, tmp, Nb, C, C, Hl, Wl, C, H, W, 3, 2, 1, 1, 1.0f);
    launch_gn(stream, tmp, gnsum2, d0_g, d0_be, hbuf, Nb, C, H * W);

    launch_conv(stream, hbuf, skip, d1_w, d1_b, tmp, Nb, 2 * C, C, H, W, C, H, W, 3, 1, 1, 0, 1.0f);
    launch_gn(stream, tmp, gnsum3, d1_g, d1_be, ybuf, Nb, C, H * W);

    launch_conv(stream, ybuf, nullptr, ro_w, ro_b, xrec, Nb, C, C, H, W, 4, H, W, 1, 1, 0, 0, 1.0f);

    vq_finalize_kernel<<<1, 256, 0, stream>>>(counts, lossA, out, out + 1 + Nb * 4 * H * W,
                                              ncodes, 1.0f / (float)rows,
                                              0.25f / (float)((size_t)rows * C));
}
